// PatchAutoencoder_35811437314304
// MI455X (gfx1250) — matrix operand order repaired and checked by reading, not compiled, not run on hardware
//
#include <hip/hip_runtime.h>

typedef __attribute__((ext_vector_type(16))) _Float16 v16h;
typedef __attribute__((ext_vector_type(8)))  _Float16 v8h;
typedef __attribute__((ext_vector_type(4)))  _Float16 v4h;
typedef __attribute__((ext_vector_type(8)))  float    v8f;
typedef __attribute__((ext_vector_type(4)))  float    v4f;

#define IMG     224
#define PSZ     16
#define STR     8
#define EDIM    128
#define NPATCH  27                 // (224-16)/8 + 1
#define LPATCH  (NPATCH * NPATCH)  // 729
#define KPIX    256                // 16*16
#define MTILE   64
#define NTILESL 12                 // ceil(729/64)
#define BATCH   64

__global__ void zero_f32(float* __restrict__ p, int n) {
  int i = blockIdx.x * blockDim.x + threadIdx.x;
  int stride = gridDim.x * blockDim.x;
  for (; i < n; i += stride) p[i] = 0.0f;
}

// 1-D overlap coverage count for stride-8 / size-16 patches on a 224 axis.
__device__ __forceinline__ float cov1d(int p) {
  int hh = p >> 3; if (hh > NPATCH - 1) hh = NPATCH - 1;
  int ll = (p >= PSZ) ? (((p - PSZ) >> 3) + 1) : 0;
  return (float)(hh - ll + 1);
}

__global__ void normalize_f32(float* __restrict__ out, int n) {
  int i = blockIdx.x * blockDim.x + threadIdx.x;
  int stride = gridDim.x * blockDim.x;
  for (; i < n; i += stride) {
    int pix = i % (IMG * IMG);
    int h = pix / IMG, w = pix % IMG;
    out[i] = out[i] / (cov1d(h) * cov1d(w));
  }
}

// A-fragment (16x32 f16): one contiguous 16-half chunk per lane (k = 16*hi + s).
__device__ __forceinline__ v16h load_fragA(const _Float16* base, int hi) {
  return *(const v16h*)(base + hi * 16);
}

// Convert a 32768-element f32 weight panel to f16 in LDS (coalesced).
__device__ __forceinline__ void stage_weights(const float* __restrict__ g,
                                              _Float16* __restrict__ s, int tid) {
  #pragma unroll 8
  for (int i = 0; i < 32; ++i) {
    int idx = (tid + i * 256) * 4;          // 1024 floats per sweep
    v4f f = *(const v4f*)(g + idx);
    v4h h;
    h.x = (_Float16)f.x; h.y = (_Float16)f.y;
    h.z = (_Float16)f.z; h.w = (_Float16)f.w;
    *(v4h*)(s + idx) = h;
  }
}

__global__ void __launch_bounds__(256)
patch_autoenc_wmma(const float* __restrict__ x,
                   const float* __restrict__ Wenc,   // [3][128][256]
                   const float* __restrict__ benc,   // [3][128]
                   const float* __restrict__ Wdec,   // [3][256][128]
                   const float* __restrict__ bdec,   // [3][256]
                   float* __restrict__ out) {
  // 112 KB static LDS (320 KB/WGP available on CDNA5)
  __shared__ __attribute__((aligned(32))) _Float16 sW[EDIM * KPIX];   // 64 KB, reused enc->dec
  __shared__ __attribute__((aligned(32))) _Float16 sP[MTILE * KPIX];  // 32 KB patch tile
  __shared__ __attribute__((aligned(32))) _Float16 sEmb[MTILE * EDIM];// 16 KB emb tile

  const int tileL = blockIdx.x;  // 0..11
  const int c     = blockIdx.y;  // 0..2
  const int b     = blockIdx.z;  // 0..63
  const int tid   = threadIdx.x;
  const int lane  = tid & 31;
  const int wv    = tid >> 5;    // 8 waves
  const int lo    = lane & 15;
  const int hi    = lane >> 4;

  const float* xc  = x    + (size_t)(b * 3 + c) * (IMG * IMG);
  const float* WeC = Wenc + (size_t)c * (EDIM * KPIX);
  const float* WdC = Wdec + (size_t)c * (KPIX * EDIM);

  // ---- stage encoder weights (f32 -> f16) and patch tile into LDS ----
  stage_weights(WeC, sW, tid);
  {
    int lLoc = tid >> 2;          // 0..63 : patch row within tile
    int q    = tid & 3;           // quarter of the 256-wide row
    int l    = tileL * MTILE + lLoc;
    bool valid = l < LPATCH;
    int ph = 0, pw = 0;
    if (valid) { ph = l / NPATCH; pw = l % NPATCH; }
    #pragma unroll 4
    for (int j4 = 0; j4 < 16; ++j4) {
      int k = q * 64 + j4 * 4;          // 4 contiguous pixels (16B aligned)
      v4h hv = {};
      if (valid) {
        int h  = ph * STR + (k >> 4);
        int wc = pw * STR + (k & 15);
        v4f f = *(const v4f*)(xc + h * IMG + wc);
        hv.x = (_Float16)f.x; hv.y = (_Float16)f.y;
        hv.z = (_Float16)f.z; hv.w = (_Float16)f.w;
      }
      *(v4h*)(sP + lLoc * KPIX + k) = hv;
    }
  }
  __syncthreads();

  // ---- Encoder GEMM: emb[64x128] = sP[64x256] x Wenc^T, f32 accum ----
  {
    const int mRow  = wv & 3;     // 4 M-subtiles of 16
    const int nHalf = wv >> 2;    // 2 N-halves of 64
    v8f acc[4] = {};
    #pragma unroll
    for (int ks = 0; ks < 8; ++ks) {
      const int K0 = ks * 32;
      v16h a = load_fragA(sP + (mRow * 16 + lo) * KPIX + K0, hi);
      #pragma unroll
      for (int t = 0; t < 4; ++t) {
        const int n = nHalf * 64 + t * 16;
        v16h bf = *(const v16h*)(sW + (n + lo) * KPIX + K0 + hi * 16);
        acc[t] = __builtin_amdgcn_wmma_f32_16x16x32_f16(
            false, a, false, bf, (short)0, acc[t], false, false);
      }
    }
    // bias + store emb tile to LDS (f16) for the decoder GEMM
    #pragma unroll
    for (int t = 0; t < 4; ++t) {
      const int n = nHalf * 64 + t * 16 + lo;
      const float bias = benc[c * EDIM + n];
      #pragma unroll
      for (int r = 0; r < 8; ++r) {
        const int m = mRow * 16 + hi * 8 + r;
        sEmb[m * EDIM + n] = (_Float16)(acc[t][r] + bias);
      }
    }
  }
  __syncthreads();

  // ---- swap weight panel: decoder weights into the same LDS buffer ----
  stage_weights(WdC, sW, tid);
  __syncthreads();

  // ---- Decoder GEMM: rec[64x256] = sEmb[64x128] x Wdec^T, + scatter ----
  {
    const int mRow = wv & 3;      // 4 M-subtiles of 16
    const int nGrp = wv >> 2;     // 2 N-groups of 128
    v8f acc[8] = {};
    #pragma unroll
    for (int es = 0; es < 4; ++es) {
      const int E0 = es * 32;
      v16h a = load_fragA(sEmb + (mRow * 16 + lo) * EDIM + E0, hi);
      #pragma unroll
      for (int t = 0; t < 8; ++t) {
        const int n = nGrp * 128 + t * 16;
        v16h bf = *(const v16h*)(sW + (n + lo) * EDIM + E0 + hi * 16);
        acc[t] = __builtin_amdgcn_wmma_f32_16x16x32_f16(
            false, a, false, bf, (short)0, acc[t], false, false);
      }
    }
    // bias + scatter-add reconstructed patches (overlaps -> f32 atomics)
    float* outc = out + (size_t)(b * 3 + c) * (IMG * IMG);
    #pragma unroll
    for (int t = 0; t < 8; ++t) {
      const int kpix = nGrp * 128 + t * 16 + lo;   // pixel index in patch
      const float bias = bdec[c * KPIX + kpix];
      const int kr = kpix >> 4, kc = kpix & 15;
      #pragma unroll
      for (int r = 0; r < 8; ++r) {
        const int m = mRow * 16 + hi * 8 + r;
        const int l = tileL * MTILE + m;
        if (l < LPATCH) {
          const int ph = l / NPATCH, pw = l % NPATCH;
          atomicAdd(&outc[(ph * STR + kr) * IMG + (pw * STR + kc)],
                    acc[t][r] + bias);
        }
      }
    }
  }
}

extern "C" void kernel_launch(void* const* d_in, const int* in_sizes, int n_in,
                              void* d_out, int out_size, void* d_ws, size_t ws_size,
                              hipStream_t stream) {
  (void)in_sizes; (void)n_in; (void)d_ws; (void)ws_size;
  const float* x  = (const float*)d_in[0];
  const float* We = (const float*)d_in[1];
  const float* be = (const float*)d_in[2];
  const float* Wd = (const float*)d_in[3];
  const float* bd = (const float*)d_in[4];
  float* out = (float*)d_out;

  const int total = BATCH * 3 * IMG * IMG;   // == out_size

  zero_f32<<<4096, 256, 0, stream>>>(out, total);

  dim3 grid(NTILESL, 3, BATCH);              // 12 x 3 x 64 = 2304 WGs
  patch_autoenc_wmma<<<grid, 256, 0, stream>>>(x, We, be, Wd, bd, out);

  normalize_f32<<<4096, 256, 0, stream>>>(out, total);
}
